// AttentionLayer_3367254360497
// MI455X (gfx1250) — matrix operand order repaired and checked by reading, not compiled, not run on hardware
//
#include <hip/hip_runtime.h>

// ---------------- types ----------------
typedef __bf16 bf16;
typedef __attribute__((ext_vector_type(4)))  __bf16 bf16x4;
typedef __attribute__((ext_vector_type(8)))  __bf16 bf16x8;
typedef __attribute__((ext_vector_type(16))) __bf16 v16bf;
typedef __attribute__((ext_vector_type(8)))  float  v8f;
typedef __attribute__((ext_vector_type(4)))  float  f32x4;

#define D_MODEL 1024
#define N_HEADS 16
#define HEAD_E  64
#define BATCH   2
#define SEQ     2048

#ifndef __has_builtin
#define __has_builtin(x) 0
#endif

#if __has_builtin(__builtin_amdgcn_global_load_async_to_lds_b128)
#define HAS_ASYNC 1
typedef int v4i_vs __attribute__((vector_size(16)));
typedef __attribute__((address_space(1))) v4i_vs gv4i;
typedef __attribute__((address_space(3))) v4i_vs lv4i;
static __device__ __forceinline__ void async_copy16(const bf16* g, bf16* l) {
  __builtin_amdgcn_global_load_async_to_lds_b128((gv4i*)g, (lv4i*)l, 0, 0);
}
static __device__ __forceinline__ void wait_async0() {
#if __has_builtin(__builtin_amdgcn_s_wait_asynccnt)
  __builtin_amdgcn_s_wait_asynccnt(0);
#else
  asm volatile("s_wait_asynccnt 0x0" ::: "memory");
#endif
}
#else
#define HAS_ASYNC 0
#endif

static __device__ __forceinline__ v16bf cmb(bf16x8 lo, bf16x8 hi) {
  return __builtin_shufflevector(lo, hi, 0,1,2,3,4,5,6,7,8,9,10,11,12,13,14,15);
}
static __device__ __forceinline__ v8f wmma_bf16(v16bf a, v16bf b, v8f c) {
  return __builtin_amdgcn_wmma_f32_16x16x32_bf16(false, a, false, b, (short)0, c, false, false);
}

// VALU lane-xor max within rows of 16 via DPP16 ROW_XMASK (no LDS, no waits).
template <int CTRL>
static __device__ __forceinline__ float dppmax(float x) {
  const int t = __builtin_amdgcn_update_dpp(0, __float_as_int(x), CTRL, 0xf, 0xf, true);
  return fmaxf(x, __int_as_float(t));
}

// =====================================================================
// GEMM: out[M,1024] = (A[M,1024] @ W[1024,1024] + bias) * oscale
// Double-buffered LDS, 128x128 block tile, 8 waves, wave tile 64x32.
// =====================================================================
template <bool A_IS_F32, bool OUT_IS_F32>
__global__ __launch_bounds__(256) void gemm_bias_kernel(
    const void* __restrict__ Ap, const float* __restrict__ W,
    const float* __restrict__ bias, void* __restrict__ Outp, float oscale) {
  __shared__ bf16 As[2][128][40];  // A tile row-major, stride 80B
  __shared__ bf16 Wt[2][128][40];  // W tile transposed [n][k], stride 80B

  const int tid  = threadIdx.x;
  const int lane = tid & 31;
  const int w    = tid >> 5;
  const int wm   = w >> 2;
  const int wn   = w & 3;
  const int lsel = lane >> 4;
  const int lrow = lane & 15;

  const int rbase = blockIdx.x * 128;
  const int cbase = blockIdx.y * 128;

  v8f acc[4][2];
#pragma unroll
  for (int i = 0; i < 4; ++i)
#pragma unroll
    for (int j = 0; j < 2; ++j)
#pragma unroll
      for (int r = 0; r < 8; ++r) acc[i][j][r] = 0.0f;

  const int quad = tid & 7;         // A staging col group
  const int r0   = tid >> 3;        // A staging row 0..31
  const int col  = (tid & 31) * 4;  // W staging col 0..124
  const int rrb  = tid >> 5;        // W staging k row 0..7

  // ---- prologue: stage tile kt=0 into buffer 0 ----
#pragma unroll
  for (int i = 0; i < 4; ++i) {
    const int r = r0 + i * 32;
    bf16x4 b;
    if (A_IS_F32) {
      f32x4 v = *reinterpret_cast<const f32x4*>(
          (const float*)Ap + (size_t)(rbase + r) * 1024 + quad * 4);
      b.x = (bf16)v.x; b.y = (bf16)v.y; b.z = (bf16)v.z; b.w = (bf16)v.w;
    } else {
      b = *reinterpret_cast<const bf16x4*>(
          (const bf16*)Ap + (size_t)(rbase + r) * 1024 + quad * 4);
    }
    *reinterpret_cast<bf16x4*>(&As[0][r][quad * 4]) = b;
  }
#pragma unroll
  for (int i = 0; i < 4; ++i) {
    const int rr = rrb + i * 8;
    f32x4 v = *reinterpret_cast<const f32x4*>(W + (size_t)rr * 1024 + cbase + col);
    Wt[0][col + 0][rr] = (bf16)v.x;
    Wt[0][col + 1][rr] = (bf16)v.y;
    Wt[0][col + 2][rr] = (bf16)v.z;
    Wt[0][col + 3][rr] = (bf16)v.w;
  }
  __syncthreads();

#pragma unroll 1
  for (int kt = 0; kt < 32; ++kt) {
    const int cur  = kt & 1;
    const int nxt  = cur ^ 1;
    const int k1   = (kt + 1) * 32;
    const bool has_next = (kt < 31);

    // ---- issue global loads for next tile (overlap with WMMA) ----
    f32x4  aregf[4];
    bf16x4 aregb[4];
    f32x4  wreg[4];
    if (has_next) {
#pragma unroll
      for (int i = 0; i < 4; ++i) {
        const int r = r0 + i * 32;
        if (A_IS_F32)
          aregf[i] = *reinterpret_cast<const f32x4*>(
              (const float*)Ap + (size_t)(rbase + r) * 1024 + k1 + quad * 4);
        else
          aregb[i] = *reinterpret_cast<const bf16x4*>(
              (const bf16*)Ap + (size_t)(rbase + r) * 1024 + k1 + quad * 4);
      }
#pragma unroll
      for (int i = 0; i < 4; ++i) {
        const int rr = rrb + i * 8;
        wreg[i] = *reinterpret_cast<const f32x4*>(
            W + (size_t)(k1 + rr) * 1024 + cbase + col);
      }
      if (kt < 30)  // prefetch tile kt+2 (global_prefetch path)
        __builtin_prefetch(W + (size_t)(k1 + 32 + rrb) * 1024 + cbase + col, 0, 1);
    }

    // ---- fragments + 8 WMMAs from current buffer ----
    v16bf afrag[4];
#pragma unroll
    for (int mt = 0; mt < 4; ++mt) {
      const int m = wm * 64 + mt * 16 + lrow;
      bf16x8 lo = *reinterpret_cast<const bf16x8*>(&As[cur][m][lsel * 8]);
      bf16x8 hi = *reinterpret_cast<const bf16x8*>(&As[cur][m][16 + lsel * 8]);
      afrag[mt] = cmb(lo, hi);
    }
    v16bf bfrag[2];
#pragma unroll
    for (int nt = 0; nt < 2; ++nt) {
      const int n = wn * 32 + nt * 16 + lrow;
      bf16x8 lo = *reinterpret_cast<const bf16x8*>(&Wt[cur][n][lsel * 8]);
      bf16x8 hi = *reinterpret_cast<const bf16x8*>(&Wt[cur][n][16 + lsel * 8]);
      bfrag[nt] = cmb(lo, hi);
    }
#pragma unroll
    for (int mt = 0; mt < 4; ++mt)
#pragma unroll
      for (int nt = 0; nt < 2; ++nt)
        acc[mt][nt] = wmma_bf16(afrag[mt], bfrag[nt], acc[mt][nt]);

    // ---- commit next tile into other buffer ----
    if (has_next) {
#pragma unroll
      for (int i = 0; i < 4; ++i) {
        const int r = r0 + i * 32;
        bf16x4 b;
        if (A_IS_F32) {
          b.x = (bf16)aregf[i].x; b.y = (bf16)aregf[i].y;
          b.z = (bf16)aregf[i].z; b.w = (bf16)aregf[i].w;
        } else {
          b = aregb[i];
        }
        *reinterpret_cast<bf16x4*>(&As[nxt][r][quad * 4]) = b;
      }
#pragma unroll
      for (int i = 0; i < 4; ++i) {
        const int rr = rrb + i * 8;
        Wt[nxt][col + 0][rr] = (bf16)wreg[i].x;
        Wt[nxt][col + 1][rr] = (bf16)wreg[i].y;
        Wt[nxt][col + 2][rr] = (bf16)wreg[i].z;
        Wt[nxt][col + 3][rr] = (bf16)wreg[i].w;
      }
    }
    __syncthreads();
  }

  // ---- epilogue ----
#pragma unroll
  for (int mt = 0; mt < 4; ++mt)
#pragma unroll
    for (int nt = 0; nt < 2; ++nt) {
      const int n = cbase + wn * 32 + nt * 16 + lrow;
      const float bv = bias[n];
#pragma unroll
      for (int r = 0; r < 8; ++r) {
        const int m = rbase + wm * 64 + mt * 16 + r + 8 * lsel;
        const float val = (acc[mt][nt][r] + bv) * oscale;
        if (OUT_IS_F32)
          ((float*)Outp)[(size_t)m * 1024 + n] = val;
        else
          ((bf16*)Outp)[(size_t)m * 1024 + n] = (bf16)val;
      }
    }
}

// =====================================================================
// Flash attention: block = (128 q rows, head, batch); wave owns 16 rows.
// S streamed in blocks of 64; double-buffered k/v tiles; k via async LDS.
// q was pre-scaled by 1/sqrt(E) in its projection epilogue.
// Row-sum of P via WMMA against an all-ones B (broadcast, no shuffles).
// Row-max via DPP16 ROW_XMASK butterfly (pure VALU, no LDS waits).
// =====================================================================
__global__ __launch_bounds__(256) void attn_kernel(
    const bf16* __restrict__ qb, const bf16* __restrict__ kb,
    const bf16* __restrict__ vb, bf16* __restrict__ ob) {
  __shared__ bf16 ksh[2][64][72];    // k tile [s][e], stride 144B
  __shared__ bf16 vT[2][64][72];     // v tile transposed [e][s], stride 144B
  __shared__ bf16 Pbuf[8][16][64];   // per-wave P bounce, stride 128B

  const int tid  = threadIdx.x;
  const int lane = tid & 31;
  const int w    = tid >> 5;
  const int lsel = lane >> 4;
  const int lrow = lane & 15;

  const int hc    = blockIdx.y * HEAD_E;
  const size_t bo = (size_t)blockIdx.z * SEQ * D_MODEL;
  const bf16* qp = qb + bo;
  const bf16* kp = kb + bo;
  const bf16* vp = vb + bo;
  bf16*       op = ob + bo;

  const int qrow0 = blockIdx.x * 128 + w * 16;

  // resident q fragments (16x64 as two K=32 chunks)
  v16bf qa[2];
#pragma unroll
  for (int c = 0; c < 2; ++c) {
    const bf16* base = qp + (size_t)(qrow0 + lrow) * 1024 + hc + c * 32 + lsel * 8;
    bf16x8 lo = *reinterpret_cast<const bf16x8*>(base);
    bf16x8 hi = *reinterpret_cast<const bf16x8*>(base + 16);
    qa[c] = cmb(lo, hi);
  }
  v16bf onesf;
#pragma unroll
  for (int i = 0; i < 16; ++i) onesf[i] = (bf16)1.0f;

  float mrow[8];
  v8f accO[4], lacc;
#pragma unroll
  for (int r = 0; r < 8; ++r) { mrow[r] = -1e30f; lacc[r] = 0.0f; }
#pragma unroll
  for (int j = 0; j < 4; ++j)
#pragma unroll
    for (int r = 0; r < 8; ++r) accO[j][r] = 0.0f;

  // staging map: thread -> (s row 0..63, e group of 16)
  const int ss = tid & 63;
  const int e0 = (tid >> 6) * 16;

  // ---- prologue: stage S-block 0 into buffer 0 ----
  {
    const bf16* ks = kp + (size_t)ss * 1024 + hc + e0;
    const bf16* vs = vp + (size_t)ss * 1024 + hc + e0;
#if HAS_ASYNC
    async_copy16(ks, &ksh[0][ss][e0]);
    async_copy16(ks + 8, &ksh[0][ss][e0 + 8]);
#else
    *reinterpret_cast<bf16x8*>(&ksh[0][ss][e0]) =
        *reinterpret_cast<const bf16x8*>(ks);
    *reinterpret_cast<bf16x8*>(&ksh[0][ss][e0 + 8]) =
        *reinterpret_cast<const bf16x8*>(ks + 8);
#endif
    bf16x8 v0 = *reinterpret_cast<const bf16x8*>(vs);
    bf16x8 v1 = *reinterpret_cast<const bf16x8*>(vs + 8);
#pragma unroll
    for (int j = 0; j < 8; ++j) vT[0][e0 + j][ss] = v0[j];
#pragma unroll
    for (int j = 0; j < 8; ++j) vT[0][e0 + 8 + j][ss] = v1[j];
  }
#if HAS_ASYNC
  wait_async0();
#endif
  __syncthreads();

#pragma unroll 1
  for (int sb = 0; sb < SEQ / 64; ++sb) {
    const int buf  = sb & 1;
    const int nbuf = buf ^ 1;
    const bool has_next = (sb + 1 < SEQ / 64);

    // ---- prefetch next S-block: k async -> LDS, v global -> regs ----
    bf16x8 v0, v1;
    if (has_next) {
      const int sn = (sb + 1) * 64 + ss;
      const bf16* ks = kp + (size_t)sn * 1024 + hc + e0;
      const bf16* vs = vp + (size_t)sn * 1024 + hc + e0;
#if HAS_ASYNC
      async_copy16(ks, &ksh[nbuf][ss][e0]);
      async_copy16(ks + 8, &ksh[nbuf][ss][e0 + 8]);
#else
      v0 = *reinterpret_cast<const bf16x8*>(ks);
      *reinterpret_cast<bf16x8*>(&ksh[nbuf][ss][e0]) = v0;
      v0 = *reinterpret_cast<const bf16x8*>(ks + 8);
      *reinterpret_cast<bf16x8*>(&ksh[nbuf][ss][e0 + 8]) = v0;
#endif
      v0 = *reinterpret_cast<const bf16x8*>(vs);
      v1 = *reinterpret_cast<const bf16x8*>(vs + 8);
    }

    // ---- scores: 4 s-tiles x (K=64 -> 2 WMMA) ----
    v8f sc[4];
#pragma unroll
    for (int t = 0; t < 4; ++t) {
      const bf16* kbase = &ksh[buf][t * 16 + lrow][0];
      bf16x8 lo = *reinterpret_cast<const bf16x8*>(kbase + lsel * 8);
      bf16x8 hi = *reinterpret_cast<const bf16x8*>(kbase + 16 + lsel * 8);
      v16bf kf0 = cmb(lo, hi);
      lo = *reinterpret_cast<const bf16x8*>(kbase + 32 + lsel * 8);
      hi = *reinterpret_cast<const bf16x8*>(kbase + 48 + lsel * 8);
      v16bf kf1 = cmb(lo, hi);
      v8f c = {0.f, 0.f, 0.f, 0.f, 0.f, 0.f, 0.f, 0.f};
      c = wmma_bf16(qa[0], kf0, c);
      c = wmma_bf16(qa[1], kf1, c);
      sc[t] = c;
    }

    // ---- online softmax: DPP16 ROW_XMASK butterfly for row max ----
    float mx[8];
#pragma unroll
    for (int r = 0; r < 8; ++r)
      mx[r] = fmaxf(fmaxf(sc[0][r], sc[1][r]), fmaxf(sc[2][r], sc[3][r]));
#pragma unroll
    for (int r = 0; r < 8; ++r) mx[r] = dppmax<0x168>(mx[r]);  // xor 8
#pragma unroll
    for (int r = 0; r < 8; ++r) mx[r] = dppmax<0x164>(mx[r]);  // xor 4
#pragma unroll
    for (int r = 0; r < 8; ++r) mx[r] = dppmax<0x162>(mx[r]);  // xor 2
#pragma unroll
    for (int r = 0; r < 8; ++r) mx[r] = dppmax<0x161>(mx[r]);  // xor 1

    float alpha[8];
#pragma unroll
    for (int r = 0; r < 8; ++r) {
      const float mnew = fmaxf(mrow[r], mx[r]);
      alpha[r] = __expf(mrow[r] - mnew);
      mrow[r]  = mnew;
#pragma unroll
      for (int t = 0; t < 4; ++t) sc[t][r] = __expf(sc[t][r] - mnew);
    }
#pragma unroll
    for (int j = 0; j < 4; ++j)
#pragma unroll
      for (int r = 0; r < 8; ++r) accO[j][r] *= alpha[r];
#pragma unroll
    for (int r = 0; r < 8; ++r) lacc[r] *= alpha[r];

    // ---- P: C-layout -> A-layout via per-wave LDS bounce ----
#pragma unroll
    for (int t = 0; t < 4; ++t)
#pragma unroll
      for (int r = 0; r < 8; ++r)
        Pbuf[w][r + 8 * lsel][lrow + 16 * t] = (bf16)sc[t][r];

    v16bf pa[2];
#pragma unroll
    for (int c = 0; c < 2; ++c) {
      bf16x8 lo = *reinterpret_cast<const bf16x8*>(&Pbuf[w][lrow][c * 32 + lsel * 8]);
      bf16x8 hi = *reinterpret_cast<const bf16x8*>(&Pbuf[w][lrow][c * 32 + 16 + lsel * 8]);
      pa[c] = cmb(lo, hi);
    }

    // ---- row sums via WMMA against ones (broadcast to all lanes) ----
    lacc = wmma_bf16(pa[0], onesf, lacc);
    lacc = wmma_bf16(pa[1], onesf, lacc);

    // ---- O += P @ V (4 e-tiles x 2 K-chunks) ----
#pragma unroll
    for (int j = 0; j < 4; ++j) {
      const bf16* vbase = &vT[buf][j * 16 + lrow][0];
#pragma unroll
      for (int c = 0; c < 2; ++c) {
        bf16x8 lo = *reinterpret_cast<const bf16x8*>(vbase + c * 32 + lsel * 8);
        bf16x8 hi = *reinterpret_cast<const bf16x8*>(vbase + c * 32 + 16 + lsel * 8);
        accO[j] = wmma_bf16(pa[c], cmb(lo, hi), accO[j]);
      }
    }

    // ---- commit prefetched v into other buffer ----
    if (has_next) {
#pragma unroll
      for (int j = 0; j < 8; ++j) vT[nbuf][e0 + j][ss] = v0[j];
#pragma unroll
      for (int j = 0; j < 8; ++j) vT[nbuf][e0 + 8 + j][ss] = v1[j];
    }
#if HAS_ASYNC
    wait_async0();
#endif
    __syncthreads();
  }

  // ---- epilogue: normalize, store bf16 ----
#pragma unroll
  for (int r = 0; r < 8; ++r) {
    const float inv = 1.0f / lacc[r];
    const int row = qrow0 + r + 8 * lsel;
#pragma unroll
    for (int j = 0; j < 4; ++j)
      op[(size_t)row * 1024 + hc + j * 16 + lrow] = (bf16)(accO[j][r] * inv);
  }
}

// =====================================================================
// launch
// =====================================================================
extern "C" void kernel_launch(void* const* d_in, const int* in_sizes, int n_in,
                              void* d_out, int out_size, void* d_ws,
                              size_t ws_size, hipStream_t stream) {
  const float* Q  = (const float*)d_in[0];
  const float* K  = (const float*)d_in[1];
  const float* V  = (const float*)d_in[2];
  const float* Wq = (const float*)d_in[3];
  const float* bq = (const float*)d_in[4];
  const float* Wk = (const float*)d_in[5];
  const float* bk = (const float*)d_in[6];
  const float* Wv = (const float*)d_in[7];
  const float* bv = (const float*)d_in[8];
  const float* We = (const float*)d_in[9];
  const float* be = (const float*)d_in[10];

  const size_t per = (size_t)BATCH * SEQ * D_MODEL;
  bf16* qb = (bf16*)d_ws;
  bf16* kb = qb + per;
  bf16* vb = qb + 2 * per;
  bf16* ob = qb + 3 * per;

  dim3 blk(256);
  dim3 gg(32, 8);  // (M/128, N/128) for M=4096, N=1024
  const float qscale = 0.125f;  // 1/sqrt(E), folded into q projection

  gemm_bias_kernel<true, false><<<gg, blk, 0, stream>>>(Q, Wq, bq, qb, qscale);
  gemm_bias_kernel<true, false><<<gg, blk, 0, stream>>>(K, Wk, bk, kb, 1.0f);
  gemm_bias_kernel<true, false><<<gg, blk, 0, stream>>>(V, Wv, bv, vb, 1.0f);

  attn_kernel<<<dim3(SEQ / 128, N_HEADS, BATCH), blk, 0, stream>>>(qb, kb, vb, ob);

  gemm_bias_kernel<false, true><<<gg, blk, 0, stream>>>(ob, We, be, d_out, 1.0f);
}
